// Multi_Head_CrossAttention_4973572129269
// MI455X (gfx1250) — matrix operand order repaired and checked by reading, not compiled, not run on hardware
//
#include <hip/hip_runtime.h>
#include <hip/hip_bf16.h>
#include <math.h>
#include <stdint.h>

typedef _Float16 half_t;
typedef __attribute__((ext_vector_type(8)))  _Float16 v8h;
typedef __attribute__((ext_vector_type(16))) _Float16 v16h;
typedef __attribute__((ext_vector_type(8)))  float    v8f;

#define DIM   256
#define HEADS 8
#define DHEAD 32
#define NB    32          // neighbors
#define DFF   512
#define ROWS  8192        // B*M
#define EPS   1e-5f
#define SCALE 0.17677669529663687f  // 1/sqrt(32)

// ---------------- WMMA helpers ----------------
__device__ __forceinline__ v16h frag_cat(v8h lo, v8h hi) {
  return __builtin_shufflevector(lo, hi, 0,1,2,3,4,5,6,7,8,9,10,11,12,13,14,15);
}

// A fragment: 16 rows x 32 K from row-major half array (leading dim ld).
__device__ __forceinline__ v16h load_a(const half_t* p, int ld, int lane, int kb) {
  int row = lane & 15;
  int k0  = kb + ((lane >> 4) << 3);
  const half_t* q = p + row * ld + k0;
  v8h lo = *(const v8h*)(q);
  v8h hi = *(const v8h*)(q + 16);
  return frag_cat(lo, hi);
}

// B fragment: W is (O x I) row-major half; out tile obase, K slice kb.
__device__ __forceinline__ v16h load_b(const half_t* W, int ldi, int lane, int obase, int kb) {
  int o    = obase + (lane & 15);
  int koff = kb + ((lane >> 4) << 3);
  const half_t* q = W + (size_t)o * ldi + koff;
  v8h lo = *(const v8h*)(q);
  v8h hi = *(const v8h*)(q + 16);
  return frag_cat(lo, hi);
}

__device__ __forceinline__ v8f wmma_f16(v16h a, v16h b, v8f c) {
  return __builtin_amdgcn_wmma_f32_16x16x32_f16(false, a, false, b, (short)0, c, false, false);
}

// ---------------- prep: f32 -> f16 ----------------
__global__ void k_cvt(const float* __restrict__ s, half_t* __restrict__ d, int n) {
  int i = blockIdx.x * 256 + threadIdx.x;
  if (i < n) d[i] = (half_t)s[i];
}

// ---------------- Q projection: LN(x) @ Wq^T + bq (32 rows/block) ----------------
__global__ __launch_bounds__(256) void k_qproj(
    const float* __restrict__ x, const float* __restrict__ lnw, const float* __restrict__ lnb,
    const half_t* __restrict__ Wh, const float* __restrict__ bias, float* __restrict__ qout)
{
  __shared__ half_t Abuf[32 * DIM];
  __shared__ float red[2][16][16];
  const int t = threadIdx.x, rowbase = blockIdx.x * 32;
  const int r16 = t >> 4, c16 = t & 15;
  #pragma unroll
  for (int p = 0; p < 2; ++p) {
    const float* xr = x + (size_t)(rowbase + p * 16 + r16) * DIM + c16 * 16;
    float vals[16], s = 0.f, ss = 0.f;
    #pragma unroll
    for (int i = 0; i < 16; ++i) { float v = xr[i]; vals[i] = v; s += v; ss += v * v; }
    red[0][r16][c16] = s; red[1][r16][c16] = ss;
    __syncthreads();
    s = 0.f; ss = 0.f;
    #pragma unroll
    for (int i = 0; i < 16; ++i) { s += red[0][r16][i]; ss += red[1][r16][i]; }
    float mu = s * (1.f / DIM), var = ss * (1.f / DIM) - mu * mu;
    float rstd = rsqrtf(var + EPS);
    #pragma unroll
    for (int i = 0; i < 16; ++i) {
      int c = c16 * 16 + i;
      Abuf[(p * 16 + r16) * DIM + c] = (half_t)((vals[i] - mu) * rstd * lnw[c] + lnb[c]);
    }
    __syncthreads();
  }
  const int wave = t >> 5, lane = t & 31;
  const int obase0 = wave * 32, obase1 = obase0 + 16;
  v8f z = {0.f,0.f,0.f,0.f,0.f,0.f,0.f,0.f};
  v8f acc[2][2] = {{z, z}, {z, z}};
  #pragma unroll
  for (int ks = 0; ks < 8; ++ks) {
    v16h b0 = load_b(Wh, DIM, lane, obase0, ks * 32);
    v16h b1 = load_b(Wh, DIM, lane, obase1, ks * 32);
    #pragma unroll
    for (int rt = 0; rt < 2; ++rt) {
      v16h a = load_a(Abuf + rt * 16 * DIM, DIM, lane, ks * 32);
      acc[0][rt] = wmma_f16(a, b0, acc[0][rt]);
      acc[1][rt] = wmma_f16(a, b1, acc[1][rt]);
    }
  }
  #pragma unroll
  for (int o = 0; o < 2; ++o) {
    int col = (o ? obase1 : obase0) + (lane & 15);
    float bb = bias[col];
    #pragma unroll
    for (int rt = 0; rt < 2; ++rt)
      #pragma unroll
      for (int j = 0; j < 8; ++j) {
        int rr = rt * 16 + j + ((lane >> 4) << 3);
        qout[(size_t)(rowbase + rr) * DIM + col] = acc[o][rt][j] + bb;
      }
  }
}

// ---------------- fused K/V projection + per-node attention (2 nodes/block) ----------------
// dynamic LDS layout (bytes):
#define KH_OFF 0                        // 64x256 f16  = 32768
#define VH_OFF 32768                    // 64x256 f16  = 32768
#define UB_OFF 65536                    // ybuf 64x256 f32 (64KB), later reused as kfh+vfh (f16)
#define RD_OFF 131072                   // 512 f32 = 2048
#define QS_OFF 133120                   // 512 f32 = 2048
#define AT_OFF 135168                   // 512 f32 = 2048
#define SM_TOT 137216

#define ASYNC_CP(OFFSTR)                                                     \
  asm volatile("global_load_async_to_lds_b128 %0, %1, off offset:" OFFSTR    \
               :: "v"(ldsa), "v"(ga) : "memory")

__global__ __launch_bounds__(256) void k_attn(
    const float* __restrict__ y,
    const float* __restrict__ lnkw, const float* __restrict__ lnkb,
    const half_t* __restrict__ Wkh, const float* __restrict__ bk,
    const float* __restrict__ lnvw, const float* __restrict__ lnvb,
    const half_t* __restrict__ Wvh, const float* __restrict__ bv,
    const float* __restrict__ qg, half_t* __restrict__ ah)
{
  extern __shared__ char smem[];
  half_t* kh  = (half_t*)(smem + KH_OFF);
  half_t* vh  = (half_t*)(smem + VH_OFF);
  float*  ybuf= (float*)(smem + UB_OFF);
  half_t* kfh = (half_t*)(smem + UB_OFF);          // aliases ybuf (dead after LN)
  half_t* vfh = (half_t*)(smem + UB_OFF + 32768);
  float*  rd  = (float*)(smem + RD_OFF);
  float*  qs  = (float*)(smem + QS_OFF);
  float*  at  = (float*)(smem + AT_OFF);

  const int t = threadIdx.x;
  const size_t idx0 = (size_t)blockIdx.x * 2;

  // ---- async DMA: 64KB of y (2 nodes) -> LDS ybuf ----
  {
    unsigned int ldsa = (unsigned int)(uintptr_t)ybuf + t * 16;
    unsigned long long ga =
        (unsigned long long)(uintptr_t)(y + idx0 * NB * DIM) + (unsigned long long)t * 16;
    ASYNC_CP("0");     ASYNC_CP("4096");  ASYNC_CP("8192");  ASYNC_CP("12288");
    ASYNC_CP("16384"); ASYNC_CP("20480"); ASYNC_CP("24576"); ASYNC_CP("28672");
    ASYNC_CP("32768"); ASYNC_CP("36864"); ASYNC_CP("40960"); ASYNC_CP("45056");
    ASYNC_CP("49152"); ASYNC_CP("53248"); ASYNC_CP("57344"); ASYNC_CP("61440");
    qs[t]       = qg[idx0 * DIM + t];
    qs[256 + t] = qg[idx0 * DIM + 256 + t];
    asm volatile("s_wait_asynccnt 0x0" ::: "memory");
  }
  __syncthreads();

  // ---- LN over 64 y-rows (shared mu/sigma, two affines) ----
  {
    const int r = t >> 2, c4 = t & 3;   // 4 threads/row, 64 cols each
    float s = 0.f, ss = 0.f;
    #pragma unroll
    for (int i = 0; i < 64; ++i) { float v = ybuf[r * DIM + c4 * 64 + i]; s += v; ss += v * v; }
    rd[r * 4 + c4] = s; rd[256 + r * 4 + c4] = ss;
    __syncthreads();
    s = 0.f; ss = 0.f;
    #pragma unroll
    for (int i = 0; i < 4; ++i) { s += rd[r * 4 + i]; ss += rd[256 + r * 4 + i]; }
    float mu = s * (1.f / DIM), var = ss * (1.f / DIM) - mu * mu;
    float rstd = rsqrtf(var + EPS);
    #pragma unroll
    for (int i = 0; i < 64; ++i) {
      int c = c4 * 64 + i;
      float nv = (ybuf[r * DIM + c] - mu) * rstd;
      kh[r * DIM + c] = (half_t)(nv * lnkw[c] + lnkb[c]);
      vh[r * DIM + c] = (half_t)(nv * lnvw[c] + lnvb[c]);
    }
  }
  __syncthreads();   // ybuf dead; kfh/vfh region may now be written

  // ---- K and V projections: wave owns 2 out-tiles, B reused across 4 row-tiles ----
  const int wave = t >> 5, lane = t & 31;
  #pragma unroll
  for (int m = 0; m < 2; ++m) {
    const half_t* Ab = m ? vh : kh;
    const half_t* W  = m ? Wvh : Wkh;
    const float*  bs = m ? bv : bk;
    half_t*       Ob = m ? vfh : kfh;
    const int obase0 = wave * 32, obase1 = obase0 + 16;
    v8f z = {0.f,0.f,0.f,0.f,0.f,0.f,0.f,0.f};
    v8f acc[2][4] = {{z, z, z, z}, {z, z, z, z}};
    #pragma unroll
    for (int ks = 0; ks < 8; ++ks) {
      v16h b0 = load_b(W, DIM, lane, obase0, ks * 32);
      v16h b1 = load_b(W, DIM, lane, obase1, ks * 32);
      #pragma unroll
      for (int rt = 0; rt < 4; ++rt) {
        v16h a = load_a(Ab + rt * 16 * DIM, DIM, lane, ks * 32);
        acc[0][rt] = wmma_f16(a, b0, acc[0][rt]);
        acc[1][rt] = wmma_f16(a, b1, acc[1][rt]);
      }
    }
    #pragma unroll
    for (int o = 0; o < 2; ++o) {
      int col = (o ? obase1 : obase0) + (lane & 15);
      float bb = bs[col];
      #pragma unroll
      for (int rt = 0; rt < 4; ++rt)
        #pragma unroll
        for (int j = 0; j < 8; ++j) {
          int rr = rt * 16 + j + ((lane >> 4) << 3);
          Ob[rr * DIM + col] = (half_t)(acc[o][rt][j] + bb);
        }
    }
  }
  __syncthreads();

  // ---- attention: wave = head, lane = neighbor (then lane = d); 2 nodes ----
  #pragma unroll
  for (int g = 0; g < 2; ++g) {
    const int h = wave, n = lane;
    const int rbase = g * 32;
    float lg = 0.f;
    #pragma unroll
    for (int d = 0; d < DHEAD; ++d)
      lg += qs[g * DIM + h * DHEAD + d] * (float)kfh[(rbase + n) * DIM + h * DHEAD + d];
    lg *= SCALE;
    float mx = lg;
    #pragma unroll
    for (int off = 16; off > 0; off >>= 1) mx = fmaxf(mx, __shfl_xor(mx, off, 32));
    float e = expf(lg - mx);
    float sm = e;
    #pragma unroll
    for (int off = 16; off > 0; off >>= 1) sm += __shfl_xor(sm, off, 32);
    at[(g * 8 + h) * NB + n] = e / sm;     // same-wave LDS ops are in-order
    const int d = lane;
    float av = 0.f;
    #pragma unroll
    for (int nn = 0; nn < NB; ++nn)
      av += at[(g * 8 + h) * NB + nn] * (float)vfh[(rbase + nn) * DIM + h * DHEAD + d];
    ah[(idx0 + g) * DIM + h * DHEAD + d] = (half_t)av;
  }
}

// ---------------- out projection + pre-LN (32 rows/block) ----------------
__global__ __launch_bounds__(256) void k_pproj(
    const half_t* __restrict__ ahp, const half_t* __restrict__ Wph, const float* __restrict__ bp,
    const float* __restrict__ prew, const float* __restrict__ preb,
    float* __restrict__ zln, half_t* __restrict__ zlnh)
{
  __shared__ float zbuf[32 * DIM];
  __shared__ float red[2][16][16];
  const int t = threadIdx.x, rowbase = blockIdx.x * 32;
  const int wave = t >> 5, lane = t & 31;
  const half_t* Ag = ahp + (size_t)rowbase * DIM;
  const int obase0 = wave * 32, obase1 = obase0 + 16;
  v8f z = {0.f,0.f,0.f,0.f,0.f,0.f,0.f,0.f};
  v8f acc[2][2] = {{z, z}, {z, z}};
  #pragma unroll
  for (int ks = 0; ks < 8; ++ks) {
    v16h b0 = load_b(Wph, DIM, lane, obase0, ks * 32);
    v16h b1 = load_b(Wph, DIM, lane, obase1, ks * 32);
    #pragma unroll
    for (int rt = 0; rt < 2; ++rt) {
      v16h a = load_a(Ag + rt * 16 * DIM, DIM, lane, ks * 32);
      acc[0][rt] = wmma_f16(a, b0, acc[0][rt]);
      acc[1][rt] = wmma_f16(a, b1, acc[1][rt]);
    }
  }
  #pragma unroll
  for (int o = 0; o < 2; ++o) {
    int col = (o ? obase1 : obase0) + (lane & 15);
    float bb = bp[col];
    #pragma unroll
    for (int rt = 0; rt < 2; ++rt)
      #pragma unroll
      for (int j = 0; j < 8; ++j) {
        int rr = rt * 16 + j + ((lane >> 4) << 3);
        zbuf[rr * DIM + col] = acc[o][rt][j] + bb;
      }
  }
  __syncthreads();
  const int r16 = t >> 4, c16 = t & 15;
  #pragma unroll
  for (int p = 0; p < 2; ++p) {
    float vals[16], s = 0.f, ss = 0.f;
    #pragma unroll
    for (int i = 0; i < 16; ++i) {
      float v = zbuf[(p * 16 + r16) * DIM + c16 * 16 + i];
      vals[i] = v; s += v; ss += v * v;
    }
    red[0][r16][c16] = s; red[1][r16][c16] = ss;
    __syncthreads();
    s = 0.f; ss = 0.f;
    #pragma unroll
    for (int i = 0; i < 16; ++i) { s += red[0][r16][i]; ss += red[1][r16][i]; }
    float mu = s * (1.f / DIM), var = ss * (1.f / DIM) - mu * mu;
    float rstd = rsqrtf(var + EPS);
    #pragma unroll
    for (int i = 0; i < 16; ++i) {
      int c = c16 * 16 + i;
      float nv = (vals[i] - mu) * rstd * prew[c] + preb[c];
      size_t o = (size_t)(rowbase + p * 16 + r16) * DIM + c;
      zln[o] = nv; zlnh[o] = (half_t)nv;
    }
    __syncthreads();
  }
}

// ---------------- FFN up + GELU (32 rows/block) ----------------
__global__ __launch_bounds__(256) void k_ffn1(
    const half_t* __restrict__ zlnh, const half_t* __restrict__ W1h, const float* __restrict__ b1,
    half_t* __restrict__ h1h)
{
  const int t = threadIdx.x, rowbase = blockIdx.x * 32;
  const int wave = t >> 5, lane = t & 31;
  const half_t* Ag = zlnh + (size_t)rowbase * DIM;
  #pragma unroll
  for (int pass = 0; pass < 2; ++pass) {
    const int obase0 = (pass * 16 + wave * 2) * 16, obase1 = obase0 + 16;
    v8f z = {0.f,0.f,0.f,0.f,0.f,0.f,0.f,0.f};
    v8f acc[2][2] = {{z, z}, {z, z}};
    #pragma unroll
    for (int ks = 0; ks < 8; ++ks) {
      v16h b0 = load_b(W1h, DIM, lane, obase0, ks * 32);
      v16h b1 = load_b(W1h, DIM, lane, obase1, ks * 32);
      #pragma unroll
      for (int rt = 0; rt < 2; ++rt) {
        v16h a = load_a(Ag + rt * 16 * DIM, DIM, lane, ks * 32);
        acc[0][rt] = wmma_f16(a, b0, acc[0][rt]);
        acc[1][rt] = wmma_f16(a, b1, acc[1][rt]);
      }
    }
    #pragma unroll
    for (int o = 0; o < 2; ++o) {
      int col = (o ? obase1 : obase0) + (lane & 15);
      float bb = b1[col];
      #pragma unroll
      for (int rt = 0; rt < 2; ++rt)
        #pragma unroll
        for (int j = 0; j < 8; ++j) {
          int rr = rt * 16 + j + ((lane >> 4) << 3);
          float v = acc[o][rt][j] + bb;
          float g = 0.5f * v * (1.f + erff(v * 0.7071067811865475f));
          h1h[(size_t)(rowbase + rr) * DFF + col] = (half_t)g;
        }
    }
  }
}

// ---------------- FFN down + residual + post-LN (32 rows/block) ----------------
__global__ __launch_bounds__(256) void k_ffn2(
    const half_t* __restrict__ h1h, const half_t* __restrict__ W2h, const float* __restrict__ b2,
    const float* __restrict__ zln, const float* __restrict__ postw, const float* __restrict__ postb,
    float* __restrict__ out)
{
  __shared__ float zbuf[32 * DIM];
  __shared__ float red[2][16][16];
  const int t = threadIdx.x, rowbase = blockIdx.x * 32;
  const int wave = t >> 5, lane = t & 31;
  const half_t* Ag = h1h + (size_t)rowbase * DFF;
  const int obase0 = wave * 32, obase1 = obase0 + 16;
  v8f z = {0.f,0.f,0.f,0.f,0.f,0.f,0.f,0.f};
  v8f acc[2][2] = {{z, z}, {z, z}};
  #pragma unroll
  for (int ks = 0; ks < 16; ++ks) {
    v16h b0 = load_b(W2h, DFF, lane, obase0, ks * 32);
    v16h b1 = load_b(W2h, DFF, lane, obase1, ks * 32);
    #pragma unroll
    for (int rt = 0; rt < 2; ++rt) {
      v16h a = load_a(Ag + rt * 16 * DFF, DFF, lane, ks * 32);
      acc[0][rt] = wmma_f16(a, b0, acc[0][rt]);
      acc[1][rt] = wmma_f16(a, b1, acc[1][rt]);
    }
  }
  #pragma unroll
  for (int o = 0; o < 2; ++o) {
    int col = (o ? obase1 : obase0) + (lane & 15);
    float bb = b2[col];
    #pragma unroll
    for (int rt = 0; rt < 2; ++rt)
      #pragma unroll
      for (int j = 0; j < 8; ++j) {
        int rr = rt * 16 + j + ((lane >> 4) << 3);
        zbuf[rr * DIM + col] = acc[o][rt][j] + bb;
      }
  }
  __syncthreads();
  const int r16 = t >> 4, c16 = t & 15;
  #pragma unroll
  for (int p = 0; p < 2; ++p) {
    float vals[16], s = 0.f, ss = 0.f;
    #pragma unroll
    for (int i = 0; i < 16; ++i) {
      int c = c16 * 16 + i;
      float v = zbuf[(p * 16 + r16) * DIM + c] +
                zln[(size_t)(rowbase + p * 16 + r16) * DIM + c];
      vals[i] = v; s += v; ss += v * v;
    }
    red[0][r16][c16] = s; red[1][r16][c16] = ss;
    __syncthreads();
    s = 0.f; ss = 0.f;
    #pragma unroll
    for (int i = 0; i < 16; ++i) { s += red[0][r16][i]; ss += red[1][r16][i]; }
    float mu = s * (1.f / DIM), var = ss * (1.f / DIM) - mu * mu;
    float rstd = rsqrtf(var + EPS);
    #pragma unroll
    for (int i = 0; i < 16; ++i) {
      int c = c16 * 16 + i;
      out[(size_t)(rowbase + p * 16 + r16) * DIM + c] =
          (vals[i] - mu) * rstd * postw[c] + postb[c];
    }
    __syncthreads();
  }
}

// ---------------- host ----------------
extern "C" void kernel_launch(void* const* d_in, const int* in_sizes, int n_in,
                              void* d_out, int out_size, void* d_ws, size_t ws_size,
                              hipStream_t stream) {
  const float* x    = (const float*)d_in[0];
  const float* y    = (const float*)d_in[1];
  const float* lnqw = (const float*)d_in[2];
  const float* lnqb = (const float*)d_in[3];
  const float* Wq   = (const float*)d_in[4];
  const float* bq   = (const float*)d_in[5];
  const float* lnkw = (const float*)d_in[6];
  const float* lnkb = (const float*)d_in[7];
  const float* Wk   = (const float*)d_in[8];
  const float* bk   = (const float*)d_in[9];
  const float* lnvw = (const float*)d_in[10];
  const float* lnvb = (const float*)d_in[11];
  const float* Wv   = (const float*)d_in[12];
  const float* bv   = (const float*)d_in[13];
  const float* Wp   = (const float*)d_in[14];
  const float* bp   = (const float*)d_in[15];
  const float* prew = (const float*)d_in[16];
  const float* preb = (const float*)d_in[17];
  const float* W1   = (const float*)d_in[18];
  const float* b1   = (const float*)d_in[19];
  const float* W2   = (const float*)d_in[20];
  const float* b2   = (const float*)d_in[21];
  const float* postw= (const float*)d_in[22];
  const float* postb= (const float*)d_in[23];
  float* out = (float*)d_out;

  char* base = (char*)d_ws;
  size_t off = 0;
  auto take = [&](size_t bytes) { char* p = base + off; off += (bytes + 255) & ~(size_t)255; return p; };
  half_t* Wqh  = (half_t*)take(DIM * DIM * 2);
  half_t* Wkh  = (half_t*)take(DIM * DIM * 2);
  half_t* Wvh  = (half_t*)take(DIM * DIM * 2);
  half_t* Wph  = (half_t*)take(DIM * DIM * 2);
  half_t* W1h  = (half_t*)take(DFF * DIM * 2);
  half_t* W2h  = (half_t*)take(DIM * DFF * 2);
  float*  q    = (float*)take((size_t)ROWS * DIM * 4);
  half_t* ahp  = (half_t*)take((size_t)ROWS * DIM * 2);
  float*  zln  = (float*)take((size_t)ROWS * DIM * 4);
  half_t* zlnh = (half_t*)take((size_t)ROWS * DIM * 2);
  half_t* h1h  = (half_t*)take((size_t)ROWS * DFF * 2);

  k_cvt<<<(DIM * DIM + 255) / 256, 256, 0, stream>>>(Wq, Wqh, DIM * DIM);
  k_cvt<<<(DIM * DIM + 255) / 256, 256, 0, stream>>>(Wk, Wkh, DIM * DIM);
  k_cvt<<<(DIM * DIM + 255) / 256, 256, 0, stream>>>(Wv, Wvh, DIM * DIM);
  k_cvt<<<(DIM * DIM + 255) / 256, 256, 0, stream>>>(Wp, Wph, DIM * DIM);
  k_cvt<<<(DFF * DIM + 255) / 256, 256, 0, stream>>>(W1, W1h, DFF * DIM);
  k_cvt<<<(DIM * DFF + 255) / 256, 256, 0, stream>>>(W2, W2h, DIM * DFF);

  k_qproj<<<ROWS / 32, 256, 0, stream>>>(x, lnqw, lnqb, Wqh, bq, q);

  static bool attr_set = false;
  if (!attr_set) {
    (void)hipFuncSetAttribute((const void*)k_attn,
                              hipFuncAttributeMaxDynamicSharedMemorySize, SM_TOT);
    attr_set = true;
  }
  k_attn<<<ROWS / 2, 256, SM_TOT, stream>>>(y, lnkw, lnkb, Wkh, bk, lnvw, lnvb, Wvh, bv, q, ahp);

  k_pproj<<<ROWS / 32, 256, 0, stream>>>(ahp, Wph, bp, prew, preb, zln, zlnh);
  k_ffn1<<<ROWS / 32, 256, 0, stream>>>(zlnh, W1h, b1, h1h);
  k_ffn2<<<ROWS / 32, 256, 0, stream>>>(h1h, W2h, b2, zln, postw, postb, out);
}
